// GatedDeltaNet_36558761623579
// MI455X (gfx1250) — compile-verified, matrix operand order checked by reading, not run on hardware
//
#include <hip/hip_runtime.h>
#include <hip/hip_bf16.h>

// Gated DeltaNet for MI455X (gfx1250, wave32).
// GEMMs run on v_wmma_f32_16x16x32_bf16 using split-bf16 (hi+lo) 3-WMMA
// accumulation for ~f32 accuracy; the delta-rule scan is a per-(b,h)
// register-resident 64x64 state recurrence.

#define B_  2
#define T_  4096
#define D_  1024
#define H_  16
#define HD_ 64
#define KC_ 4

typedef __attribute__((ext_vector_type(16))) __bf16 bf16x16;
typedef __attribute__((ext_vector_type(8)))  __bf16 bf16x8;
typedef __attribute__((ext_vector_type(8)))  float  f32x8;

// ---------------------------------------------------------------------------
// f32 -> (hi, lo) bf16 split:  x ~= hi + lo,  hi = rn_bf16(x), lo = rn_bf16(x-hi)
// ---------------------------------------------------------------------------
__global__ void gdn_cvt_split(const float* __restrict__ src, void* hi_v, void* lo_v, int n) {
    __bf16* __restrict__ hi = (__bf16*)hi_v;
    __bf16* __restrict__ lo = (__bf16*)lo_v;
    int i = blockIdx.x * blockDim.x + threadIdx.x;
    if (i < n) {
        float f = src[i];
        __bf16 h = (__bf16)f;
        hi[i] = h;
        lo[i] = (__bf16)(f - (float)h);
    }
}

// ---------------------------------------------------------------------------
// C[n,d] = sum_c A[n,c] * W[d,c]   (i.e. A @ W^T), A,W given as bf16 hi/lo.
// Wave: 16x64 output strip. Block: 8 waves -> 128x64 tile. K multiple of 32.
// A fragment (16x32 bf16): lane m (m<16) holds row M=m, K {0..7, 16..23};
// lane m+16 holds K {8..15, 24..31}  (ISA 7.12.2). B fragment mirrors with
// lane = output column, rows of W supplying B^T directly.
// ---------------------------------------------------------------------------
__device__ __forceinline__ bf16x16 gdn_load_frag(const __bf16* p0, const __bf16* p1) {
    bf16x8 a = *(const bf16x8*)p0;
    bf16x8 b = *(const bf16x8*)p1;
    bf16x16 r;
#pragma unroll
    for (int i = 0; i < 8; ++i) { r[i] = a[i]; r[i + 8] = b[i]; }
    return r;
}

__global__ __launch_bounds__(256)
void gdn_gemm_bf16split(const void* Ahi_v, const void* Alo_v,
                        const void* Whi_v, const void* Wlo_v,
                        float* __restrict__ C, int N, int K, int Dout) {
    const __bf16* __restrict__ Ahi = (const __bf16*)Ahi_v;
    const __bf16* __restrict__ Alo = (const __bf16*)Alo_v;
    const __bf16* __restrict__ Whi = (const __bf16*)Whi_v;
    const __bf16* __restrict__ Wlo = (const __bf16*)Wlo_v;

    const int wave = threadIdx.x >> 5;
    const int lane = threadIdx.x & 31;
    const int m16  = lane & 15;
    const int half = lane >> 4;

    const int mBase = blockIdx.x * 128 + wave * 16;
    const int nBase = blockIdx.y * 64;

    f32x8 acc[4] = {};

    const __bf16* arow_hi = Ahi + (size_t)(mBase + m16) * K;
    const __bf16* arow_lo = Alo + (size_t)(mBase + m16) * K;

    for (int kk = 0; kk < K; kk += 32) {
        const int o0 = kk + half * 8;
        const int o1 = kk + 16 + half * 8;
        bf16x16 ahi = gdn_load_frag(arow_hi + o0, arow_hi + o1);
        bf16x16 alo = gdn_load_frag(arow_lo + o0, arow_lo + o1);
#pragma unroll
        for (int nt = 0; nt < 4; ++nt) {
            const __bf16* wrow_hi = Whi + (size_t)(nBase + nt * 16 + m16) * K;
            const __bf16* wrow_lo = Wlo + (size_t)(nBase + nt * 16 + m16) * K;
            bf16x16 bhi = gdn_load_frag(wrow_hi + o0, wrow_hi + o1);
            bf16x16 blo = gdn_load_frag(wrow_lo + o0, wrow_lo + o1);
            acc[nt] = __builtin_amdgcn_wmma_f32_16x16x32_bf16(
                false, ahi, false, bhi, (short)0, acc[nt], false, false);
            acc[nt] = __builtin_amdgcn_wmma_f32_16x16x32_bf16(
                false, ahi, false, blo, (short)0, acc[nt], false, false);
            acc[nt] = __builtin_amdgcn_wmma_f32_16x16x32_bf16(
                false, alo, false, bhi, (short)0, acc[nt], false, false);
        }
    }
    // C/D layout: VGPR r -> (M = r + 8*half, N = m16) per 16x16 subtile.
#pragma unroll
    for (int nt = 0; nt < 4; ++nt) {
        const int col = nBase + nt * 16 + m16;
#pragma unroll
        for (int r = 0; r < 8; ++r) {
            const int row = mBase + r + half * 8;
            C[(size_t)row * Dout + col] = acc[nt][r];
        }
    }
}

// ---------------------------------------------------------------------------
// Depthwise causal conv (K=4, left pad 3):  out[t,d] = b[d] + sum_j u[t-3+j,d]*w[d,j]
// ---------------------------------------------------------------------------
__global__ void gdn_dwconv4(const float* __restrict__ raw, const float* __restrict__ cw,
                            const float* __restrict__ cb, float* __restrict__ out) {
    size_t i = (size_t)blockIdx.x * blockDim.x + threadIdx.x;
    const size_t total = (size_t)B_ * T_ * D_;
    if (i >= total) return;
    const int d = (int)(i % D_);
    const int t = (int)((i / D_) % T_);
    float acc = cb[d];
#pragma unroll
    for (int j = 0; j < KC_; ++j) {
        const int ts = t - (KC_ - 1) + j;
        if (ts >= 0) acc += raw[i - (size_t)(KC_ - 1 - j) * D_] * cw[d * KC_ + j];
    }
    out[i] = acc;
}

// ---------------------------------------------------------------------------
// L2 normalize k rows of length HD=64 (one 64-thread block per (b,t,h) row)
// ---------------------------------------------------------------------------
__global__ __launch_bounds__(64)
void gdn_knorm(float* __restrict__ k) {
    __shared__ float red[64];
    const size_t base = (size_t)blockIdx.x * HD_;
    const int tid = threadIdx.x;
    float val = k[base + tid];
    red[tid] = val * val;
    __syncthreads();
    for (int s = 32; s > 0; s >>= 1) {
        if (tid < s) red[tid] += red[tid + s];
        __syncthreads();
    }
    const float scale = 1.0f / fmaxf(sqrtf(red[0]), 1e-12f);
    k[base + tid] = val * scale;
}

// ---------------------------------------------------------------------------
// Gates: decay = sigmoid(x.Wg_h) (= exp(log_sigmoid)), beta = sigmoid(x.Wb_h)
// One block per token row n; 128 threads = 32 dots (16 Wg + 16 Wb) x 4 partials.
// ---------------------------------------------------------------------------
__global__ __launch_bounds__(128)
void gdn_gates(const float* __restrict__ x, const float* __restrict__ Wg,
               const float* __restrict__ Wb, float* __restrict__ decay,
               float* __restrict__ beta) {
    __shared__ float xs[D_];
    __shared__ float red[128];
    const int n = blockIdx.x;
    const int tid = threadIdx.x;
    for (int c = tid; c < D_; c += 128) xs[c] = x[(size_t)n * D_ + c];
    __syncthreads();
    const int outIdx = tid >> 2;   // 0..31
    const int part   = tid & 3;
    const float* Wrow = (outIdx < H_) ? (Wg + (size_t)outIdx * D_)
                                      : (Wb + (size_t)(outIdx - H_) * D_);
    float s = 0.f;
    const int c0 = part * (D_ / 4);
    for (int c = c0; c < c0 + D_ / 4; ++c) s += xs[c] * Wrow[c];
    red[tid] = s;
    __syncthreads();
    if (part == 0) {
        const float z  = red[tid] + red[tid + 1] + red[tid + 2] + red[tid + 3];
        const float sg = 1.0f / (1.0f + expf(-z));
        if (outIdx < H_) decay[(size_t)n * H_ + outIdx] = sg;
        else             beta[(size_t)n * H_ + (outIdx - H_)] = sg;
    }
}

// ---------------------------------------------------------------------------
// Delta-rule scan: one 64-thread block per (b,h); thread tid owns state
// column S[:, tid] (64 registers). k/q rows staged in LDS per step.
//   Sk[v]   = sum_k S[k,v] * kt[k]
//   S[k,v]  = e*S[k,v] + (beta*(vt[v]-Sk[v])) * kt[k]
//   o[v]    = sum_k S[k,v] * qt[k]
// ---------------------------------------------------------------------------
__global__ __launch_bounds__(64)
void gdn_scan(const float* __restrict__ q, const float* __restrict__ k,
              const float* __restrict__ v, const float* __restrict__ decay,
              const float* __restrict__ beta, float* __restrict__ o) {
    const int b = blockIdx.x / H_;
    const int h = blockIdx.x % H_;
    const int tid = threadIdx.x;      // value-dim index
    float S[HD_];
#pragma unroll
    for (int i = 0; i < HD_; ++i) S[i] = 0.f;

    __shared__ float sk[HD_], sq[HD_];

    for (int t = 0; t < T_; ++t) {
        const size_t base = ((size_t)(b * T_ + t)) * D_ + (size_t)h * HD_;
        const size_t gb   = (size_t)(b * T_ + t) * H_ + h;
        sk[tid] = k[base + tid];
        sq[tid] = q[base + tid];
        const float vt = v[base + tid];
        const float e  = decay[gb];
        const float bt = beta[gb];
        __syncthreads();

        float Sk = 0.f;
#pragma unroll
        for (int i = 0; i < HD_; ++i) Sk += S[i] * sk[i];
        const float coef = bt * (vt - Sk);
        float ot = 0.f;
#pragma unroll
        for (int i = 0; i < HD_; ++i) {
            S[i] = e * S[i] + coef * sk[i];
            ot  += S[i] * sq[i];
        }
        o[base + tid] = ot;
        __syncthreads();
    }
}

// ---------------------------------------------------------------------------
// Host launcher
// ---------------------------------------------------------------------------
extern "C" void kernel_launch(void* const* d_in, const int* in_sizes, int n_in,
                              void* d_out, int out_size, void* d_ws, size_t ws_size,
                              hipStream_t stream) {
    const float* x   = (const float*)d_in[0];
    const float* Wq  = (const float*)d_in[1];
    const float* Wk  = (const float*)d_in[2];
    const float* Wv  = (const float*)d_in[3];
    const float* Wo  = (const float*)d_in[4];
    const float* Wg  = (const float*)d_in[5];
    const float* Wb  = (const float*)d_in[6];
    const float* qcw = (const float*)d_in[7];
    const float* qcb = (const float*)d_in[8];
    const float* kcw = (const float*)d_in[9];
    const float* kcb = (const float*)d_in[10];
    const float* vcw = (const float*)d_in[11];
    const float* vcb = (const float*)d_in[12];

    const size_t N  = (size_t)B_ * T_;   // 8192 token rows
    const size_t ND = N * D_;            // 8,388,608 elements
    const size_t DD = (size_t)D_ * D_;   // 1,048,576

    // Workspace carve-up (256B aligned).
    char*  base = (char*)d_ws;
    size_t off  = 0;
    auto carve  = [&](size_t bytes) -> char* {
        char* p = base + off;
        off = (off + bytes + 255) & ~(size_t)255;
        return p;
    };
    void*  xhi   = carve(ND * 2);          // bf16
    void*  xlo   = carve(ND * 2);          // bf16
    char*  whi   = carve(4 * DD * 2);      // bf16: Wq,Wk,Wv,Wo hi
    char*  wlo   = carve(4 * DD * 2);      // bf16: lo
    float* qraw  = (float*)carve(ND * 4);
    float* kraw  = (float*)carve(ND * 4);
    float* vraw  = (float*)carve(ND * 4);
    float* qc    = (float*)carve(ND * 4);
    float* kc    = (float*)carve(ND * 4);
    float* vc    = (float*)carve(ND * 4);
    float* decay = (float*)carve(N * H_ * 4);
    float* betab = (float*)carve(N * H_ * 4);
    float* obuf  = qraw;                   // reuse: qraw dead after conv
    void*  ohi   = xhi;                    // reuse: x splits dead after GEMMs
    void*  olo   = xlo;

    auto woff = [&](char* p, int idx) -> void* { return (void*)(p + (size_t)idx * DD * 2); };

    const int cvtB = 256;
    // 1) split x and the four big weights into bf16 hi/lo
    gdn_cvt_split<<<dim3((ND + cvtB - 1) / cvtB), dim3(cvtB), 0, stream>>>(x, xhi, xlo, (int)ND);
    gdn_cvt_split<<<dim3((DD + cvtB - 1) / cvtB), dim3(cvtB), 0, stream>>>(Wq, woff(whi, 0), woff(wlo, 0), (int)DD);
    gdn_cvt_split<<<dim3((DD + cvtB - 1) / cvtB), dim3(cvtB), 0, stream>>>(Wk, woff(whi, 1), woff(wlo, 1), (int)DD);
    gdn_cvt_split<<<dim3((DD + cvtB - 1) / cvtB), dim3(cvtB), 0, stream>>>(Wv, woff(whi, 2), woff(wlo, 2), (int)DD);
    gdn_cvt_split<<<dim3((DD + cvtB - 1) / cvtB), dim3(cvtB), 0, stream>>>(Wo, woff(whi, 3), woff(wlo, 3), (int)DD);

    // 2) WMMA GEMMs: qraw/kraw/vraw = x @ W{q,k,v}^T
    dim3 ggrid((unsigned)(N / 128), (unsigned)(D_ / 64));
    gdn_gemm_bf16split<<<ggrid, dim3(256), 0, stream>>>(xhi, xlo, woff(whi, 0), woff(wlo, 0), qraw, (int)N, D_, D_);
    gdn_gemm_bf16split<<<ggrid, dim3(256), 0, stream>>>(xhi, xlo, woff(whi, 1), woff(wlo, 1), kraw, (int)N, D_, D_);
    gdn_gemm_bf16split<<<ggrid, dim3(256), 0, stream>>>(xhi, xlo, woff(whi, 2), woff(wlo, 2), vraw, (int)N, D_, D_);

    // 3) gates (decay = exp(log_sigmoid) = sigmoid, beta = sigmoid)
    gdn_gates<<<dim3((unsigned)N), dim3(128), 0, stream>>>(x, Wg, Wb, decay, betab);

    // 4) depthwise causal convs
    const size_t tot = ND;
    dim3 cgrid((unsigned)((tot + 255) / 256));
    gdn_dwconv4<<<cgrid, dim3(256), 0, stream>>>(qraw, qcw, qcb, qc);
    gdn_dwconv4<<<cgrid, dim3(256), 0, stream>>>(kraw, kcw, kcb, kc);
    gdn_dwconv4<<<cgrid, dim3(256), 0, stream>>>(vraw, vcw, vcb, vc);

    // 5) L2-normalize k per head row
    gdn_knorm<<<dim3((unsigned)(N * H_)), dim3(64), 0, stream>>>(kc);

    // 6) sequential delta-rule scan -> o (reuses qraw storage)
    gdn_scan<<<dim3(B_ * H_), dim3(64), 0, stream>>>(qc, kc, vc, decay, betab, obuf);

    // 7) output projection: out = o @ Wo^T (WMMA again)
    gdn_cvt_split<<<dim3((ND + cvtB - 1) / cvtB), dim3(cvtB), 0, stream>>>(obuf, ohi, olo, (int)ND);
    gdn_gemm_bf16split<<<ggrid, dim3(256), 0, stream>>>(ohi, olo, woff(whi, 3), woff(wlo, 3), (float*)d_out, (int)N, D_, D_);

    (void)in_sizes; (void)n_in; (void)out_size; (void)ws_size;
}
